// TopKLoRALinear_22247930594054
// MI455X (gfx1250) — matrix operand order repaired and checked by reading, not compiled, not run on hardware
//
#include <hip/hip_runtime.h>
#include <hip/hip_bf16.h>

typedef __attribute__((ext_vector_type(16))) __bf16 v16bf;
typedef __attribute__((ext_vector_type(2)))  __bf16 v2bf;
typedef __attribute__((ext_vector_type(8)))  float  v8f;
typedef __attribute__((ext_vector_type(4)))  int    v4i;

#define DIN   4096
#define DOUT  4096
#define MTOT  16384   // B_SZ * SEQ
#define RR    64
#define KTILE 32
#define BM    128
#define BN    256
#define SCALEF 2.0f   // ALPHA / R = 128/64

#if __has_builtin(__builtin_amdgcn_global_load_async_to_lds_b128)
#define HAVE_ASYNC 1
#else
#define HAVE_ASYNC 0
#endif

// ---------- helpers ----------

__device__ inline unsigned short f2bf(float f) {
    unsigned int u = __float_as_uint(f);
    u += 0x7FFFu + ((u >> 16) & 1u);   // round to nearest even
    return (unsigned short)(u >> 16);
}

// Pack two fp32 into two bf16 (RNE) with the hardware pack instruction.
__device__ inline unsigned int pack2(float a, float b) {
#if __has_builtin(__builtin_amdgcn_cvt_pk_bf16_f32)
    union { v2bf v; unsigned int u; } t;
    t.v = __builtin_amdgcn_cvt_pk_bf16_f32(a, b);
    return t.u;
#else
    unsigned int r;
    asm("v_cvt_pk_bf16_f32 %0, %1, %2" : "=v"(r) : "v"(a), "v"(b));
    return r;
#endif
}

__device__ inline void wait_async() {
#if HAVE_ASYNC
#if __has_builtin(__builtin_amdgcn_s_wait_asynccnt)
    __builtin_amdgcn_s_wait_asynccnt(0);
#else
    asm volatile("s_wait_asynccnt 0x0" ::: "memory");
#endif
#endif
}

union FragU { v16bf v; uint4 q[2]; };

// A-operand (16x32 bf16, MxK): lane holds row M=laneLo; elements 0..7 -> K=hi*8..,
// elements 8..15 -> K=16+hi*8..  -> two 16B contiguous LDS reads.
__device__ inline v16bf load_a_frag(const unsigned short* lds, int row, int hi) {
    FragU f;
    const unsigned short* p = lds + row * KTILE;
    f.q[0] = *(const uint4*)(p + hi * 8);
    f.q[1] = *(const uint4*)(p + 16 + hi * 8);
    return f.v;
}
// B-operand (32x16 bf16, KxN): lane holds col N=laneLo; elements 0..7 -> K=hi*8..,
// elements 8..15 -> K=16+hi*8..  -> two 16B contiguous LDS reads.
__device__ inline v16bf load_b_frag(const unsigned short* lds, int col, int hi) {
    FragU f;
    const unsigned short* p = lds + col * KTILE;
    f.q[0] = *(const uint4*)(p + hi * 8);
    f.q[1] = *(const uint4*)(p + 16 + hi * 8);
    return f.v;
}

// Stage a rows x 32 bf16 tile: async global->LDS (ASYNCcnt) if available.
__device__ inline void stage_bf16(unsigned short* dst, const unsigned short* src,
                                  int rows, int srcStride, int rowBase, int kk,
                                  int tid, int nthreads) {
    int chunks = rows * 4;
    for (int c = tid; c < chunks; c += nthreads) {
        int row = c >> 2;
        int ci  = (c & 3) * 8;
        const unsigned short* g = &src[(size_t)(rowBase + row) * srcStride + kk + ci];
        unsigned short* l = &dst[row * KTILE + ci];
#if HAVE_ASYNC
        __builtin_amdgcn_global_load_async_to_lds_b128(
            (__attribute__((address_space(1))) v4i*)g,
            (__attribute__((address_space(3))) v4i*)l, 0, 0);
#else
        *(uint4*)l = *(const uint4*)g;
#endif
    }
}
// Stage a rows x 32 tile from fp32, converting to bf16 in-register.
__device__ inline void stage_f32cvt(unsigned short* dst, const float* src,
                                    int rows, int srcStride, int rowBase, int kk,
                                    int tid, int nthreads) {
    int chunks = rows * 8;
    for (int c = tid; c < chunks; c += nthreads) {
        int row = c >> 3;
        int ci  = (c & 7) * 4;
        float4 f = *(const float4*)&src[(size_t)(rowBase + row) * srcStride + kk + ci];
        uint2 o;
        o.x = pack2(f.x, f.y);
        o.y = pack2(f.z, f.w);
        *(uint2*)&dst[row * KTILE + ci] = o;
    }
}

// ---------- fp32 -> bf16 bulk convert (W, A, B) ----------
__global__ __launch_bounds__(256) void cvt_f32_bf16(const float* __restrict__ in,
                                                    unsigned short* __restrict__ out,
                                                    int n4) {
    int i = blockIdx.x * 256 + threadIdx.x;
    if (i >= n4) return;
    float4 f = ((const float4*)in)[i];
    uint2 o;
    o.x = pack2(f.x, f.y);
    o.y = pack2(f.z, f.w);
    ((uint2*)out)[i] = o;
}

// ---------- z = x @ A^T, top-k(16 of 64) by |z|, store z*SCALE as bf16 ----------
__global__ __launch_bounds__(128) void zk_topk(const float* __restrict__ x,
                                               const unsigned short* __restrict__ abf,
                                               unsigned short* __restrict__ zsp) {
    __shared__ __align__(16) unsigned short lx[16 * KTILE];
    __shared__ __align__(16) unsigned short la[64 * KTILE];
    __shared__ float lz[16 * 64];

    const int tid    = threadIdx.x;
    const int lane   = tid & 31;
    const int wave   = tid >> 5;          // 4 waves, one 16x16 tile each
    const int laneLo = lane & 15;
    const int hi     = lane >> 4;
    const int mBase  = blockIdx.x * 16;

    v8f acc = {0.f, 0.f, 0.f, 0.f, 0.f, 0.f, 0.f, 0.f};

    for (int kk = 0; kk < DIN; kk += KTILE) {
        stage_f32cvt(lx, x, 16, DIN, mBase, kk, tid, 128);
        stage_bf16(la, abf, 64, DIN, 0, kk, tid, 128);
        wait_async();
        __syncthreads();
        v16bf a = load_a_frag(lx, laneLo, hi);
        v16bf b = load_b_frag(la, wave * 16 + laneLo, hi);
        acc = __builtin_amdgcn_wmma_f32_16x16x32_bf16(false, a, false, b,
                                                      (short)0, acc, false, false);
        __syncthreads();
    }

#pragma unroll
    for (int j = 0; j < 8; j++)
        lz[(hi * 8 + j) * 64 + wave * 16 + laneLo] = acc[j];
    __syncthreads();

    if (tid < 16) {
        const float* zr = &lz[tid * 64];
        unsigned long long used = 0ull;
        float thresh = 0.f;
        for (int p = 0; p < 16; p++) {           // selection -> 16th largest |z|
            float best = -1.f; int bi = 0;
            for (int r = 0; r < RR; r++) {
                if ((used >> r) & 1ull) continue;
                float av = fabsf(zr[r]);
                if (av > best) { best = av; bi = r; }
            }
            used |= (1ull << bi);
            thresh = best;
        }
        for (int r = 0; r < RR; r++) {
            float zv = zr[r];
            unsigned short o = (fabsf(zv) >= thresh) ? f2bf(zv * SCALEF)
                                                     : (unsigned short)0;
            zsp[(size_t)(mBase + tid) * RR + r] = o;
        }
    }
}

// ---------- out = x @ W^T + b + z_sparse @ B^T (scale folded into z) ----------
// Double-buffered LDS, async global->LDS staging for the bf16 operands.
__global__ __launch_bounds__(256) void fused_gemm(const float* __restrict__ x,
                                                  const unsigned short* __restrict__ wbf,
                                                  const unsigned short* __restrict__ bbf,
                                                  const unsigned short* __restrict__ zsp,
                                                  const float* __restrict__ bias,
                                                  float* __restrict__ out) {
    __shared__ __align__(16) unsigned short lx[2 * BM * KTILE]; // 16 KB
    __shared__ __align__(16) unsigned short lw[2 * BN * KTILE]; // 32 KB

    const int tid    = threadIdx.x;
    const int lane   = tid & 31;
    const int wave   = tid >> 5;        // 8 waves
    const int laneLo = lane & 15;
    const int hi     = lane >> 4;
    const int waveM  = wave & 1;        // 2 waves in M -> 2*64 = 128
    const int waveN  = wave >> 1;       // 4 waves in N -> 4*64 = 256

    const int mBase = blockIdx.y * BM;
    const int nBase = blockIdx.x * BN;

    v8f acc[4][4];
    const v8f vzero = {0.f, 0.f, 0.f, 0.f, 0.f, 0.f, 0.f, 0.f};
#pragma unroll
    for (int i = 0; i < 4; i++)
#pragma unroll
        for (int j = 0; j < 4; j++) acc[i][j] = vzero;

    const int NK1 = DIN / KTILE;            // 128 tiles over W
    const int NKT = NK1 + RR / KTILE;       // +2 tiles over [z_sparse | B]

    auto stage_tile = [&](int kt, int buf) {
        unsigned short* dlx = lx + buf * (BM * KTILE);
        unsigned short* dlw = lw + buf * (BN * KTILE);
        if (kt < NK1) {
            int kk = kt * KTILE;
            stage_f32cvt(dlx, x,   BM, DIN, mBase, kk, tid, 256);
            stage_bf16 (dlw, wbf, BN, DIN, nBase, kk, tid, 256);
        } else {
            int kk = (kt - NK1) * KTILE;
            stage_bf16 (dlx, zsp, BM, RR, mBase, kk, tid, 256);
            stage_bf16 (dlw, bbf, BN, RR, nBase, kk, tid, 256);
        }
    };

    stage_tile(0, 0);
    wait_async();
    __syncthreads();

    for (int kt = 0; kt < NKT; ++kt) {
        const int cur = kt & 1;
        if (kt + 1 < NKT) stage_tile(kt + 1, cur ^ 1);   // prefetch next tile

        const unsigned short* slx = lx + cur * (BM * KTILE);
        const unsigned short* slw = lw + cur * (BN * KTILE);

        v16bf af[4], bfv[4];
#pragma unroll
        for (int tm = 0; tm < 4; tm++)
            af[tm] = load_a_frag(slx, waveM * 64 + tm * 16 + laneLo, hi);
#pragma unroll
        for (int tn = 0; tn < 4; tn++)
            bfv[tn] = load_b_frag(slw, waveN * 64 + tn * 16 + laneLo, hi);
#pragma unroll
        for (int tm = 0; tm < 4; tm++)
#pragma unroll
            for (int tn = 0; tn < 4; tn++)
                acc[tm][tn] = __builtin_amdgcn_wmma_f32_16x16x32_bf16(
                    false, af[tm], false, bfv[tn], (short)0, acc[tm][tn], false, false);

        wait_async();        // next tile's async copies done (this wave's share)
        __syncthreads();     // everyone done computing cur + staging next
    }

    // ---- epilogue: + bias, store fp32 ----
#pragma unroll
    for (int tm = 0; tm < 4; tm++) {
#pragma unroll
        for (int tn = 0; tn < 4; tn++) {
            int n = nBase + waveN * 64 + tn * 16 + laneLo;
            float bv = bias[n];
#pragma unroll
            for (int j = 0; j < 8; j++) {
                int m = mBase + waveM * 64 + tm * 16 + hi * 8 + j;
                out[(size_t)m * DOUT + n] = acc[tm][tn][j] + bv;
            }
        }
    }
}

// ---------- launch ----------
extern "C" void kernel_launch(void* const* d_in, const int* in_sizes, int n_in,
                              void* d_out, int out_size, void* d_ws, size_t ws_size,
                              hipStream_t stream) {
    const float* x  = (const float*)d_in[0];   // (4,4096,4096)
    const float* W  = (const float*)d_in[1];   // (4096,4096)
    const float* bb = (const float*)d_in[2];   // (4096,)
    const float* A  = (const float*)d_in[3];   // (64,4096)
    const float* Bm = (const float*)d_in[4];   // (4096,64)
    // d_in[5] = k (always 16 for this problem; top-k path hardcoded)
    float* out = (float*)d_out;

    // workspace layout (~35 MB total)
    char* ws = (char*)d_ws;
    unsigned short* wbf = (unsigned short*)(ws);                             // 32 MB
    unsigned short* abf = (unsigned short*)(ws + 33554432ull);               // 512 KB
    unsigned short* bbf = (unsigned short*)(ws + 33554432ull + 524288ull);   // 512 KB
    unsigned short* zsp = (unsigned short*)(ws + 33554432ull + 1048576ull);  // 2 MB

    cvt_f32_bf16<<<16384, 256, 0, stream>>>(W,  wbf, (DOUT * DIN) / 4);
    cvt_f32_bf16<<<  256, 256, 0, stream>>>(A,  abf, (RR * DIN) / 4);
    cvt_f32_bf16<<<  256, 256, 0, stream>>>(Bm, bbf, (DOUT * RR) / 4);

    zk_topk<<<MTOT / 16, 128, 0, stream>>>(x, abf, zsp);

    dim3 grid(DOUT / BN, MTOT / BM);  // (16, 128)
    fused_gemm<<<grid, 256, 0, stream>>>(x, wbf, bbf, zsp, bb, out);
}
